// T5Encoder_25606595019048
// MI455X (gfx1250) — compile-verified, matrix operand order checked by reading, not run on hardware
//
#include <hip/hip_runtime.h>
#include <stdint.h>

// ---------------- model constants ----------------
#define DM     1024            // d_model
#define DFF    2816            // d_ff
#define NH     16              // heads
#define DKV    64              // head dim
#define NL     8               // layers
#define BB     8               // batch
#define SS     512             // seq
#define MROWS  (BB*SS)         // 4096 token rows

// Use gfx1250 async Global->LDS DMA for tile staging (pure-copy paths).
#define ASYNC_STAGE 1

typedef __attribute__((ext_vector_type(16))) __bf16 bf16x16;
typedef __attribute__((ext_vector_type(8)))  __bf16 bf16x8;
typedef __attribute__((ext_vector_type(8)))  float  floatx8;

// ---------------- WMMA fragment load from LDS ----------------
// LDS tile stored row-major [row][k] with row-stride `stride` (bf16 elems).
// 16-bit A/B layout (CDNA5 ISA 7.12.2): lanes 0-15 hold K 0..7 & 16..23,
// lanes 16-31 hold K 8..15 & 24..31, row index = lane&15.
__device__ __forceinline__ bf16x16 load_frag(const __bf16* p0, int stride, int lane) {
    const __bf16* p = p0 + (lane & 15) * stride + ((lane >> 4) << 3);
    bf16x16 r;
#pragma unroll
    for (int i = 0; i < 8; ++i) { r[i] = p[i]; r[i + 8] = p[i + 16]; }
    return r;
}

// ---------------- generic tiled WMMA GEMM (double-buffered, async-staged) ----
// C[M,N] (f32 acc) = A_bf16[M,K] * B[K,N]   (B optionally stored transposed [N,K])
// REQUIRES: M % 128 == 0, K % 32 == 0. If BT or !NGUARD: N % 128 == 0.
// Row byte strides 16B-aligned. Batched via blockIdx.z with power-of-two inner:
//   ptr += (z >> sh)*so + (z & ((1<<sh)-1))*si  per operand.
#define BM 128
#define BN 128
#define BK 32
#define LDSP 40   // padded row stride (bf16) to dodge bank conflicts (row = 80B)

template <typename TB, bool BT, typename TC, bool ADD, bool NGUARD>
__global__ __launch_bounds__(256)
void gemm_wmma(const __bf16* __restrict__ A, const TB* __restrict__ B, TC* __restrict__ C,
               int M, int N, int K, int lda, int ldb, int ldc,
               int ash, long long aso, long long asi,
               int bsh, long long bso, long long bsi,
               int csh, long long cso, long long csi) {
    static_assert(!BT || sizeof(TB) == 2, "transposed path is a bf16 copy");
    static_assert(!BT || !NGUARD, "transposed path requires N % 128 == 0");
    const int z = blockIdx.z;
    A += (long long)(z >> ash) * aso + (long long)(z & ((1 << ash) - 1)) * asi;
    B += (long long)(z >> bsh) * bso + (long long)(z & ((1 << bsh) - 1)) * bsi;
    C += (long long)(z >> csh) * cso + (long long)(z & ((1 << csh) - 1)) * csi;

    const int m0 = blockIdx.y * BM;
    const int n0 = blockIdx.x * BN;
    const int tid  = threadIdx.x;
    const int lane = tid & 31;
    const int w    = tid >> 5;           // 8 waves
    const int wm   = (w >> 1) * 32;      // 4 wave-rows
    const int wn   = (w & 1) * 64;       // 2 wave-cols

    __shared__ __align__(16) __bf16 As[2][BM * LDSP];
    __shared__ __align__(16) __bf16 Bs[2][BN * LDSP];

    // stage one BKxBM / BKxBN tile pair into LDS buffer `buf`
    auto stage = [&](int k0, int buf) {
        // A tile [m][k]: pure bf16 copy, 16B chunks, no guards
#pragma unroll
        for (int p = 0; p < 2; ++p) {
            int c   = tid + p * 256;                 // 0..511
            int mm  = c >> 2;                        // 0..127
            int kk8 = (c & 3) << 3;                  // 0,8,16,24
            const __bf16* gsrc = A + (long long)(m0 + mm) * lda + (k0 + kk8);
            __bf16* ldst = &As[buf][mm * LDSP + kk8];
#if ASYNC_STAGE
            asm volatile("global_load_async_to_lds_b128 %0, %1, off"
                         :: "v"((unsigned)(uintptr_t)ldst),
                            "v"((unsigned long long)(uintptr_t)gsrc)
                         : "memory");
#else
            *(bf16x8*)ldst = *(const bf16x8*)gsrc;
#endif
        }
        if constexpr (BT) {
            // B stored [N,K] bf16: pure copy, 16B chunks (N % 128 == 0 here)
#pragma unroll
            for (int p = 0; p < 2; ++p) {
                int c   = tid + p * 256;
                int nn  = c >> 2;
                int kk8 = (c & 3) << 3;
                const __bf16* gsrc = (const __bf16*)B + (long long)(n0 + nn) * ldb + (k0 + kk8);
                __bf16* ldst = &Bs[buf][nn * LDSP + kk8];
#if ASYNC_STAGE
                asm volatile("global_load_async_to_lds_b128 %0, %1, off"
                             :: "v"((unsigned)(uintptr_t)ldst),
                                "v"((unsigned long long)(uintptr_t)gsrc)
                             : "memory");
#else
                *(bf16x8*)ldst = *(const bf16x8*)gsrc;
#endif
            }
        } else {
            // B stored [K,N]: thread owns one LDS row (fixed n), 16 k values.
            // Coalesced-across-lanes loads, bf16 convert, one 32B LDS store.
            int nn = tid & 127;
            int kh = (tid >> 7) << 4;                // 0 or 16
            const TB* bp = B + (long long)(k0 + kh) * ldb + (n0 + nn);
            bf16x16 tv;
            if constexpr (NGUARD) {
                bool ok = (n0 + nn) < N;
#pragma unroll
                for (int kk = 0; kk < 16; ++kk) {
                    float v = ok ? (float)bp[(long long)kk * ldb] : 0.0f;
                    tv[kk] = (__bf16)v;
                }
            } else {
#pragma unroll
                for (int kk = 0; kk < 16; ++kk)
                    tv[kk] = (__bf16)(float)bp[(long long)kk * ldb];
            }
            *(bf16x16*)&Bs[buf][nn * LDSP + kh] = tv;
        }
    };

    floatx8 acc[2][4] = {};

    // pipeline prologue: stage tile 0
    stage(0, 0);
#if ASYNC_STAGE
    asm volatile("s_wait_asynccnt 0x0" ::: "memory");
#endif
    __syncthreads();

    const int niter = K / BK;
    for (int it = 0; it < niter; ++it) {
        const int cur = it & 1;
        // issue async stage of the NEXT tile into the other buffer; the DMA
        // overlaps with this iteration's fragment loads + WMMA.
        if (it + 1 < niter) stage((it + 1) * BK, cur ^ 1);

        bf16x16 af[2], bfr[4];
#pragma unroll
        for (int i = 0; i < 2; ++i)
            af[i] = load_frag(&As[cur][(wm + i * 16) * LDSP], LDSP, lane);
#pragma unroll
        for (int j = 0; j < 4; ++j)
            bfr[j] = load_frag(&Bs[cur][(wn + j * 16) * LDSP], LDSP, lane);

#pragma unroll
        for (int i = 0; i < 2; ++i)
#pragma unroll
            for (int j = 0; j < 4; ++j)
                acc[i][j] = __builtin_amdgcn_wmma_f32_16x16x32_bf16(
                    false, af[i], false, bfr[j], (short)0, acc[i][j], false, false);

#if ASYNC_STAGE
        asm volatile("s_wait_asynccnt 0x0" ::: "memory");
#endif
        __syncthreads();   // next tile staged; all frag reads of `cur` done
    }

    // C layout (7.12.2): VGPR r -> row r + 8*(lane>>4), col = lane&15
    const int cn  = lane & 15;
    const int chi = lane >> 4;
#pragma unroll
    for (int i = 0; i < 2; ++i)
#pragma unroll
        for (int j = 0; j < 4; ++j) {
            int gn = n0 + wn + j * 16 + cn;
            if (!NGUARD || gn < N) {
                long long rowb = (long long)(m0 + wm + i * 16 + chi * 8);
#pragma unroll
                for (int r = 0; r < 8; ++r) {
                    long long idx = (rowb + r) * ldc + gn;
                    float v = acc[i][j][r];
                    if constexpr (ADD) C[idx] = C[idx] + v;
                    else               C[idx] = (TC)v;
                }
            }
        }
}

// ---------------- embedding gather ----------------
__global__ __launch_bounds__(256)
void embed_k(const int* __restrict__ ids, const float* __restrict__ emb,
             float* __restrict__ x) {
    int row = blockIdx.x;
    long long src = (long long)ids[row] * DM;
    long long dst = (long long)row * DM;
#pragma unroll
    for (int j = 0; j < 4; ++j) {
        int c = threadIdx.x + j * 256;
        x[dst + c] = emb[src + c];
    }
}

// ---------------- T5 relative position bias -> [H,S,S] f32 ----------------
__global__ __launch_bounds__(256)
void relbias_k(const float* __restrict__ rel_bias, float* __restrict__ bias) {
    int idx = blockIdx.x * 256 + threadIdx.x;    // 0 .. S*S-1
    int ki = idx & (SS - 1);
    int qi = idx >> 9;
    int rp = ki - qi;                            // mem - ctx
    int bucket = (rp > 0) ? 16 : 0;
    int arp = rp < 0 ? -rp : rp;
    if (arp < 8) {
        bucket += arp;
    } else {
        // 8 + log(arp/8)/log(128/8) * 8 ; 8/log(16) = 2.8853900818
        int lg = 8 + (int)(__logf((float)arp * 0.125f) * 2.8853900818f);
        bucket += (lg < 15 ? lg : 15);
    }
#pragma unroll
    for (int h = 0; h < NH; ++h)
        bias[(long long)h * SS * SS + idx] = rel_bias[bucket * NH + h];
}

// ---------------- RMSNorm (row of 1024) ----------------
template <typename TOUT>
__global__ __launch_bounds__(256)
void rmsnorm_k(const float* __restrict__ x, const float* __restrict__ w,
               TOUT* __restrict__ out) {
    int row = blockIdx.x;
    const float* xr = x + (long long)row * DM;
    __shared__ float red[256];
    int t = threadIdx.x;
    float v[4]; float ss = 0.0f;
#pragma unroll
    for (int j = 0; j < 4; ++j) { v[j] = xr[t + j * 256]; ss += v[j] * v[j]; }
    red[t] = ss; __syncthreads();
    for (int s = 128; s > 0; s >>= 1) {
        if (t < s) red[t] += red[t + s];
        __syncthreads();
    }
    float scale = rsqrtf(red[0] * (1.0f / DM) + 1e-6f);
    TOUT* o = out + (long long)row * DM;
#pragma unroll
    for (int j = 0; j < 4; ++j)
        o[t + j * 256] = (TOUT)(w[t + j * 256] * v[j] * scale);
}

// ---------------- softmax(scores + bias) -> bf16 attn ----------------
__global__ __launch_bounds__(256)
void softmax_bias_k(const float* __restrict__ scores, const float* __restrict__ bias,
                    __bf16* __restrict__ attn) {
    int row = blockIdx.x;                 // (b*16+h)*512 + qi
    int qi = row & (SS - 1);
    int h  = (row >> 9) & (NH - 1);
    const float* srow = scores + (long long)row * SS;
    const float* brow = bias + ((long long)h * SS + qi) * SS;
    __shared__ float red[256];
    int t = threadIdx.x;
    float v0 = srow[t]       + brow[t];
    float v1 = srow[t + 256] + brow[t + 256];
    red[t] = fmaxf(v0, v1); __syncthreads();
    for (int s = 128; s > 0; s >>= 1) {
        if (t < s) red[t] = fmaxf(red[t], red[t + s]);
        __syncthreads();
    }
    float m = red[0]; __syncthreads();
    float e0 = __expf(v0 - m), e1 = __expf(v1 - m);
    red[t] = e0 + e1; __syncthreads();
    for (int s = 128; s > 0; s >>= 1) {
        if (t < s) red[t] += red[t + s];
        __syncthreads();
    }
    float inv = 1.0f / red[0];
    __bf16* arow = attn + (long long)row * SS;
    arow[t]       = (__bf16)(e0 * inv);
    arow[t + 256] = (__bf16)(e1 * inv);
}

// ---------------- tanh-GELU gate: ff = bf16(gelu(g0)*g1) ----------------
__global__ __launch_bounds__(256)
void gelu_gate_k(const float* __restrict__ g0, const float* __restrict__ g1,
                 __bf16* __restrict__ ff, long long n) {
    long long base = (long long)blockIdx.x * 1024 + threadIdx.x;
#pragma unroll
    for (int j = 0; j < 4; ++j) {
        long long i = base + j * 256;
        if (i < n) {
            float x = g0[i];
            float inner = 0.7978845608f * (x + 0.044715f * x * x * x);
            float g = 0.5f * x * (1.0f + tanhf(inner));
            ff[i] = (__bf16)(g * g1[i]);
        }
    }
}

// ---------------- host-side orchestration ----------------
extern "C" void kernel_launch(void* const* d_in, const int* in_sizes, int n_in,
                              void* d_out, int out_size, void* d_ws, size_t ws_size,
                              hipStream_t stream) {
    const int*   input_ids = (const int*)  d_in[0];
    const float* embed     = (const float*)d_in[1];
    const float* rel_bias  = (const float*)d_in[2];
    const float* wq        = (const float*)d_in[3];
    const float* wk        = (const float*)d_in[4];
    const float* wv        = (const float*)d_in[5];
    const float* wo        = (const float*)d_in[6];
    const float* ln1_w     = (const float*)d_in[7];
    const float* wi0       = (const float*)d_in[8];
    const float* wi1       = (const float*)d_in[9];
    const float* wo_ff     = (const float*)d_in[10];
    const float* ln2_w     = (const float*)d_in[11];
    const float* final_ln  = (const float*)d_in[12];
    float* out = (float*)d_out;
    char*  ws  = (char*)d_ws;

    // workspace layout (bytes, 256-aligned)
    const size_t OFF_BIAS = 0;                       // 16*512*512*4   = 16 MiB
    const size_t OFF_X    = OFF_BIAS + 16777216;     // 4096*1024*4
    const size_t OFF_H    = OFF_X    + 16777216;     // 4096*1024*2
    const size_t OFF_Q    = OFF_H    + 8388608;
    const size_t OFF_K    = OFF_Q    + 8388608;
    const size_t OFF_V    = OFF_K    + 8388608;
    const size_t OFF_CTX  = OFF_V    + 8388608;
    const size_t OFF_ATTN = OFF_CTX  + 8388608;      // 128*512*512*2 = 64 MiB (reused as ff bf16)
    const size_t OFF_BIG  = OFF_ATTN + 67108864;     // scores f32 128*512*512*4, or g0/g1
    const size_t OFF_G0   = OFF_BIG;
    const size_t OFF_G1   = OFF_BIG + 46137344;      // 4096*2816*4

    float*  bias  = (float*) (ws + OFF_BIAS);
    float*  x     = (float*) (ws + OFF_X);
    __bf16* h     = (__bf16*)(ws + OFF_H);
    __bf16* qb    = (__bf16*)(ws + OFF_Q);
    __bf16* kb    = (__bf16*)(ws + OFF_K);
    __bf16* vb    = (__bf16*)(ws + OFF_V);
    __bf16* ctxb  = (__bf16*)(ws + OFF_CTX);
    __bf16* attn  = (__bf16*)(ws + OFF_ATTN);
    __bf16* ffb   = (__bf16*)(ws + OFF_ATTN);        // reuse after attn consumed
    float*  score = (float*) (ws + OFF_BIG);
    float*  g0    = (float*) (ws + OFF_G0);
    float*  g1    = (float*) (ws + OFF_G1);

    embed_k<<<MROWS, 256, 0, stream>>>(input_ids, embed, x);
    relbias_k<<<(SS * SS) / 256, 256, 0, stream>>>(rel_bias, bias);

    const dim3 gDM (DM  / BN, MROWS / BM, 1);        // (8, 32)
    const dim3 gFF (DFF / BN, MROWS / BM, 1);        // (22, 32)
    const dim3 gSc (SS / BN, SS / BM, BB * NH);      // (4, 4, 128)
    const dim3 gCtx(1,       SS / BM, BB * NH);      // (1, 4, 128)
    const long long SBH = (long long)SS * DM;        // 524288: per-batch row stride

    for (int l = 0; l < NL; ++l) {
        const float* wql  = wq    + (size_t)l * DM * DM;
        const float* wkl  = wk    + (size_t)l * DM * DM;
        const float* wvl  = wv    + (size_t)l * DM * DM;
        const float* wol  = wo    + (size_t)l * DM * DM;
        const float* wi0l = wi0   + (size_t)l * DM * DFF;
        const float* wi1l = wi1   + (size_t)l * DM * DFF;
        const float* wofl = wo_ff + (size_t)l * DFF * DM;

        // ---- attention ----
        rmsnorm_k<__bf16><<<MROWS, 256, 0, stream>>>(x, ln1_w + l * DM, h);
        gemm_wmma<float, false, __bf16, false, false><<<gDM, 256, 0, stream>>>(
            h, wql, qb, MROWS, DM, DM, DM, DM, DM, 0,0,0, 0,0,0, 0,0,0);
        gemm_wmma<float, false, __bf16, false, false><<<gDM, 256, 0, stream>>>(
            h, wkl, kb, MROWS, DM, DM, DM, DM, DM, 0,0,0, 0,0,0, 0,0,0);
        gemm_wmma<float, false, __bf16, false, false><<<gDM, 256, 0, stream>>>(
            h, wvl, vb, MROWS, DM, DM, DM, DM, DM, 0,0,0, 0,0,0, 0,0,0);

        // scores[b,h] = Q[b,:,h*64:..] @ K^T   (K stored row-major -> B transposed)
        gemm_wmma<__bf16, true, float, false, false><<<gSc, 256, 0, stream>>>(
            qb, kb, score, SS, SS, DKV, DM, DM, SS,
            4, SBH, DKV,   4, SBH, DKV,   0, (long long)SS * SS, 0);

        softmax_bias_k<<<BB * NH * SS, 256, 0, stream>>>(score, bias, attn);

        // ctx[b,h] = attn @ V   (N=64 -> guarded instantiation)
        gemm_wmma<__bf16, false, __bf16, false, true><<<gCtx, 256, 0, stream>>>(
            attn, vb, ctxb, SS, DKV, SS, SS, DM, DM,
            0, (long long)SS * SS, 0,   4, SBH, DKV,   4, SBH, DKV);

        // x += ctx @ wo
        gemm_wmma<float, false, float, true, false><<<gDM, 256, 0, stream>>>(
            ctxb, wol, x, MROWS, DM, DM, DM, DM, DM, 0,0,0, 0,0,0, 0,0,0);

        // ---- gated-GELU FFN ----
        rmsnorm_k<__bf16><<<MROWS, 256, 0, stream>>>(x, ln2_w + l * DM, h);
        gemm_wmma<float, false, float, false, false><<<gFF, 256, 0, stream>>>(
            h, wi0l, g0, MROWS, DFF, DM, DM, DFF, DFF, 0,0,0, 0,0,0, 0,0,0);
        gemm_wmma<float, false, float, false, false><<<gFF, 256, 0, stream>>>(
            h, wi1l, g1, MROWS, DFF, DM, DM, DFF, DFF, 0,0,0, 0,0,0, 0,0,0);

        long long nff = (long long)MROWS * DFF;
        gelu_gate_k<<<(unsigned)((nff + 1023) / 1024), 256, 0, stream>>>(g0, g1, ffb, nff);

        // x += ff @ wo_ff
        gemm_wmma<float, false, float, true, false><<<gDM, 256, 0, stream>>>(
            ffb, wofl, x, MROWS, DM, DFF, DFF, DM, DM, 0,0,0, 0,0,0, 0,0,0);
    }

    rmsnorm_k<float><<<MROWS, 256, 0, stream>>>(x, final_ln, out);
    (void)in_sizes; (void)n_in; (void)out_size; (void)ws_size;
}
